// SharedAttentionProcessor_65292092834569
// MI455X (gfx1250) — matrix operand order repaired and checked by reading, not compiled, not run on hardware
//
#include <hip/hip_runtime.h>
#include <hip/hip_bf16.h>

#define B_ 4
#define N_ 1024
#define D_ 1024
#define H_ 16
#define HD_ 64
#define SCALE_ 0.125f
#define SHIFT_ 0.1f
#define EPS_ 1e-5f
#define SZ_ (B_ * H_ * N_ * HD_) /* 4194304 = 2^22 */

typedef _Float16 v16h __attribute__((ext_vector_type(16)));
typedef _Float16 h4 __attribute__((ext_vector_type(4)));
typedef float v8f __attribute__((ext_vector_type(8)));

#if defined(__has_builtin)
#if __has_builtin(__builtin_amdgcn_global_load_async_to_lds_b128)
#define HAVE_ASYNC_LDS 1
#endif
#endif

#ifdef HAVE_ASYNC_LDS
// Probe-derived signature: param0 is 'int __vector_size__(16) __device__ *'
// (AS1 global), param1 the LDS-side pointer (AS3), then imm offset + cpol.
typedef int i128v __attribute__((vector_size(16)));
typedef __attribute__((address_space(1))) i128v* gptr128;
typedef __attribute__((address_space(3))) i128v* lptr128;
__device__ inline void async_cp16(const void* g, void* l) {
  __builtin_amdgcn_global_load_async_to_lds_b128((gptr128)g, (lptr128)l, 0, 0);
}
__device__ inline void async_wait0() {
#if __has_builtin(__builtin_amdgcn_s_wait_asynccnt)
  __builtin_amdgcn_s_wait_asynccnt(0);
#else
  asm volatile("s_wait_asynccnt 0x0" ::: "memory");
#endif
}
#else
__device__ inline void async_cp16(const void* g, void* l) {
  *(uint4*)l = *(const uint4*)g;
}
__device__ inline void async_wait0() {}
#endif

union Frag16 {
  v16h v;
  _Float16 e[16];
  uint4 u[2];
};

__device__ inline v8f zero8() {
  v8f z;
  for (int i = 0; i < 8; ++i) z[i] = 0.f;
  return z;
}

// A fragment: elements 0..7 from run0 (8 contiguous f16), 8..15 from run1.
__device__ inline v16h ldsA(const _Float16* run0, const _Float16* run1) {
  Frag16 f;
  f.u[0] = *(const uint4*)run0;
  f.u[1] = *(const uint4*)run1;
  return f.v;
}
// B fragment: 16 contiguous f16 (K-contiguous at this lane's column).
__device__ inline v16h ldsB16(const _Float16* p) {
  Frag16 f;
  f.u[0] = ((const uint4*)p)[0];
  f.u[1] = ((const uint4*)p)[1];
  return f.v;
}

__device__ inline h4 cvt4(float4 x) {
  h4 r;
  r[0] = (_Float16)x.x;
  r[1] = (_Float16)x.y;
  r[2] = (_Float16)x.z;
  r[3] = (_Float16)x.w;
  return r;
}

// ---------------------------------------------------------------------------
// GEMM: [4096 x 1024] = X @ W + bias (+ residual). f16 WMMA, fp32 accum.
// mode 0: f32 out row-major (+residual). mode 1: f32 out [b,h,n,hd].
// mode 2: f16 out transposed [b,h,hd,n] (for V).
// Block = 256 threads (8 waves), tile 128x128, K chunks of 32.
// ---------------------------------------------------------------------------
__global__ __launch_bounds__(256) void gemm_f16_wmma(
    const float* __restrict__ X, const float* __restrict__ W,
    const float* __restrict__ bias, const float* __restrict__ residual,
    float* __restrict__ outF, _Float16* __restrict__ outH, int mode) {
  __shared__ __align__(16) _Float16 Xs[128][32];  // [row][k]
  __shared__ __align__(16) _Float16 Wt[128][32];  // [col][k] (transposed)

  const int tid = threadIdx.x;
  const int lane = tid & 31;
  const int w = tid >> 5;
  const int hi = (lane >> 4) & 1;
  const int l15 = lane & 15;
  const int rowBlk = blockIdx.y * 128;
  const int colBlk = blockIdx.x * 128;
  const int wm = (w >> 2) * 64;
  const int wn = (w & 3) * 32;

  v8f acc[4][2];
  for (int i = 0; i < 4; ++i)
    for (int j = 0; j < 2; ++j) acc[i][j] = zero8();

  for (int k0 = 0; k0 < D_; k0 += 32) {
    // Stage X tile [128x32]: 4 float4 loads per thread, batched waits.
    float4 xv[4], wv[4];
    for (int j = 0; j < 4; ++j) {
      int i4 = tid + j * 256;
      int r = i4 >> 3, kq = i4 & 7;
      xv[j] = *(const float4*)&X[(size_t)(rowBlk + r) * D_ + k0 + kq * 4];
    }
    for (int j = 0; j < 4; ++j) {
      int i4 = tid + j * 256;
      int c4 = i4 & 31, kk = i4 >> 5;
      wv[j] = *(const float4*)&W[(size_t)(k0 + kk) * D_ + colBlk + c4 * 4];
    }
    for (int j = 0; j < 4; ++j) {
      int i4 = tid + j * 256;
      int r = i4 >> 3, kq = i4 & 7;
      *(h4*)&Xs[r][kq * 4] = cvt4(xv[j]);
    }
    for (int j = 0; j < 4; ++j) {
      int i4 = tid + j * 256;
      int c4 = i4 & 31, kk = i4 >> 5;
      Wt[c4 * 4 + 0][kk] = (_Float16)wv[j].x;
      Wt[c4 * 4 + 1][kk] = (_Float16)wv[j].y;
      Wt[c4 * 4 + 2][kk] = (_Float16)wv[j].z;
      Wt[c4 * 4 + 3][kk] = (_Float16)wv[j].w;
    }
    if (k0 + 32 < D_) {
      __builtin_prefetch(&X[(size_t)(rowBlk + (tid >> 3)) * D_ + k0 + 32 + (tid & 7) * 4], 0, 1);
      __builtin_prefetch(&W[(size_t)(k0 + 32 + (tid >> 5)) * D_ + colBlk + (tid & 31) * 4], 0, 1);
    }
    __syncthreads();

    v16h bf[2];
    for (int nj = 0; nj < 2; ++nj)
      bf[nj] = ldsB16(&Wt[wn + nj * 16 + l15][hi ? 16 : 0]);

    for (int mi = 0; mi < 4; ++mi) {
      int r = wm + mi * 16 + l15;
      v16h af = ldsA(&Xs[r][hi ? 16 : 0], &Xs[r][(hi ? 16 : 0) + 8]);
      for (int nj = 0; nj < 2; ++nj) {
        acc[mi][nj] = __builtin_amdgcn_wmma_f32_16x16x32_f16(
            false, af, false, bf[nj], (short)0, acc[mi][nj], false, false);
      }
    }
    __syncthreads();
  }

  // Epilogue: C layout row = r + 8*hi, col = lane&15.
  for (int nj = 0; nj < 2; ++nj) {
    int colG = colBlk + wn + nj * 16 + l15;
    float bv = bias[colG];
    for (int mi = 0; mi < 4; ++mi) {
      for (int r = 0; r < 8; ++r) {
        int rowG = rowBlk + wm + mi * 16 + r + (hi ? 8 : 0);
        float vv = acc[mi][nj][r] + bv;
        int b = rowG >> 10, n = rowG & 1023;
        int h = colG >> 6, hd = colG & 63;
        if (mode == 0) {
          vv += residual[(size_t)rowG * D_ + colG];
          outF[(size_t)rowG * D_ + colG] = vv;
        } else if (mode == 1) {
          outF[((((size_t)b * H_ + h) * N_) + n) * HD_ + hd] = vv;
        } else {
          outH[(((size_t)b * H_ + h) * HD_ + hd) * N_ + n] = (_Float16)vv;
        }
      }
    }
  }
}

// ---------------------------------------------------------------------------
// AdaIN stats: per (sel, b, h, hd) mean & sqrt(unbiased var + eps) over n.
// ---------------------------------------------------------------------------
__global__ __launch_bounds__(256) void adain_stats(const float* __restrict__ q,
                                                   const float* __restrict__ k,
                                                   float* __restrict__ stats) {
  const int bh = blockIdx.x;
  const int sel = blockIdx.y;
  const float* buf = sel ? k : q;
  const int tid = threadIdx.x;
  const int ch = tid & 63;
  const int part = tid >> 6;
  __shared__ float s1[4][64];
  __shared__ float s2[4][64];
  const float* base = buf + (size_t)bh * N_ * HD_;
  float s = 0.f, ss = 0.f;
  for (int n = part; n < N_; n += 4) {
    float x = base[(size_t)n * HD_ + ch];
    s += x;
    ss += x * x;
  }
  s1[part][ch] = s;
  s2[part][ch] = ss;
  __syncthreads();
  if (tid < 64) {
    float S = s1[0][tid] + s1[1][tid] + s1[2][tid] + s1[3][tid];
    float SS = s2[0][tid] + s2[1][tid] + s2[2][tid] + s2[3][tid];
    float mean = S / (float)N_;
    float var = (SS - S * mean) / (float)(N_ - 1);
    float sd = sqrtf(var + EPS_);
    size_t idx = (((size_t)sel * B_ * H_ + bh) * HD_ + tid) * 2;
    stats[idx] = mean;
    stats[idx + 1] = sd;
  }
}

// ---------------------------------------------------------------------------
// AdaIN apply: q restyled in place (f32); k restyled into f16 [B,H,N,HD].
// ---------------------------------------------------------------------------
__global__ __launch_bounds__(256) void adain_apply(float* __restrict__ q,
                                                   const float* __restrict__ k,
                                                   _Float16* __restrict__ kh,
                                                   const float* __restrict__ stats) {
  for (size_t i = (size_t)blockIdx.x * 256 + threadIdx.x; i < 2 * (size_t)SZ_;
       i += (size_t)gridDim.x * 256) {
    int sel = (int)(i >> 22);
    size_t off = i & (size_t)(SZ_ - 1);
    int ch = (int)(off & 63);
    int h = (int)((off >> 16) & 15);
    int b = (int)(off >> 20);
    int bs = (b < B_ / 2) ? 0 : (B_ / 2);
    size_t si = (((size_t)sel * B_ * H_ + b * H_ + h) * HD_ + ch) * 2;
    size_t sis = (((size_t)sel * B_ * H_ + bs * H_ + h) * HD_ + ch) * 2;
    float mean = stats[si], sd = stats[si + 1];
    float means = stats[sis], sds = stats[sis + 1];
    if (sel == 0) {
      float x = q[off];
      q[off] = (x - mean) / sd * sds + means;
    } else {
      float x = k[off];
      kh[off] = (_Float16)((x - mean) / sd * sds + means);
    }
  }
}

// ---------------------------------------------------------------------------
// Flash attention over kv length 2N (second half = style-batch k/v, +SHIFT).
// K/V are pre-converted f16 (K: [B,H,N,HD]; V: [B,H,HD,N] transposed), so
// chunk staging is pure 16B copies -> async-to-LDS path when available.
// ---------------------------------------------------------------------------
__global__ __launch_bounds__(256) void flash_attn(const float* __restrict__ q,
                                                  const _Float16* __restrict__ kh,
                                                  const _Float16* __restrict__ vth,
                                                  float* __restrict__ attn) {
  __shared__ __align__(16) _Float16 Ks[32][64];     // [kv][hd]
  __shared__ __align__(16) _Float16 Vt[64][32];     // [hd][kv]
  __shared__ __align__(16) _Float16 Ps[8][16][32];  // per-wave P scratch

  const int tid = threadIdx.x, lane = tid & 31, w = tid >> 5;
  const int hi = (lane >> 4) & 1, l15 = lane & 15;
  const int bh = blockIdx.y;
  const int b = bh >> 4, h = bh & 15;
  const int bs = (b < B_ / 2) ? 0 : (B_ / 2);
  const int bhs = bs * H_ + h;
  const int qRow = blockIdx.x * 128 + w * 16;

  // Q fragments (A layout), softmax scale folded into the f16 conversion.
  const float* qBase = q + ((size_t)bh * N_ + qRow + l15) * HD_;
  Frag16 qf[2];
  for (int f = 0; f < 2; ++f) {
    for (int run = 0; run < 2; ++run) {
      int kk = f * 32 + hi * 16 + run * 8;
      float4 a = *(const float4*)&qBase[kk];
      float4 c = *(const float4*)&qBase[kk + 4];
      int e0 = run * 8;
      qf[f].e[e0 + 0] = (_Float16)(a.x * SCALE_);
      qf[f].e[e0 + 1] = (_Float16)(a.y * SCALE_);
      qf[f].e[e0 + 2] = (_Float16)(a.z * SCALE_);
      qf[f].e[e0 + 3] = (_Float16)(a.w * SCALE_);
      qf[f].e[e0 + 4] = (_Float16)(c.x * SCALE_);
      qf[f].e[e0 + 5] = (_Float16)(c.y * SCALE_);
      qf[f].e[e0 + 6] = (_Float16)(c.z * SCALE_);
      qf[f].e[e0 + 7] = (_Float16)(c.w * SCALE_);
    }
  }

  v8f acc[4];
  for (int t = 0; t < 4; ++t) acc[t] = zero8();
  float m[8], l[8];
  for (int r = 0; r < 8; ++r) {
    m[r] = -3.0e38f;
    l[r] = 0.f;
  }

  for (int c = 0; c < (2 * N_) / 32; ++c) {
    int kv0 = c * 32;
    int local = (kv0 < N_) ? kv0 : (kv0 - N_);
    int srcBH = (kv0 < N_) ? bh : bhs;
    float shift = (kv0 < N_) ? 0.f : SHIFT_;

    // K chunk: 4KB contiguous in global; one 16B copy per thread.
    const _Float16* kSrc = kh + ((size_t)srcBH * N_ + local) * HD_;
    async_cp16(kSrc + tid * 8, &Ks[0][0] + tid * 8);
    // V chunk: 64 rows x 64B from [hd][n]; one 16B copy per thread.
    {
      int r = tid >> 2, blk = tid & 3;
      const _Float16* vSrc = vth + ((size_t)srcBH * HD_ + r) * N_ + local + blk * 8;
      async_cp16(vSrc, &Vt[r][blk * 8]);
    }
    // Prefetch next chunk of K.
    int kv0n = kv0 + 32;
    if (kv0n < 2 * N_) {
      int localn = (kv0n < N_) ? kv0n : (kv0n - N_);
      int srcn = (kv0n < N_) ? bh : bhs;
      __builtin_prefetch(kh + ((size_t)srcn * N_ + localn) * HD_ + tid * 8, 0, 1);
    }
    async_wait0();
    __syncthreads();

    // S = Q @ K^T for two 16-kv subtiles.
    v8f s[2];
    for (int j = 0; j < 2; ++j) {
      v8f sj = zero8();
      int kvc = j * 16 + l15;
      sj = __builtin_amdgcn_wmma_f32_16x16x32_f16(
          false, qf[0].v, false, ldsB16(&Ks[kvc][hi ? 16 : 0]), (short)0, sj, false, false);
      sj = __builtin_amdgcn_wmma_f32_16x16x32_f16(
          false, qf[1].v, false, ldsB16(&Ks[kvc][32 + (hi ? 16 : 0)]), (short)0, sj, false, false);
      s[j] = sj;
    }

    // Online softmax. Row scalars are uniform within each 16-lane half.
    float p0[8], p1[8];
    for (int r = 0; r < 8; ++r) {
      float a0 = s[0][r] + shift, a1 = s[1][r] + shift;
      float mx = fmaxf(a0, a1);
      for (int d = 8; d >= 1; d >>= 1) mx = fmaxf(mx, __shfl_xor(mx, d));
      float mn = fmaxf(m[r], mx);
      float e0 = __expf(a0 - mn), e1 = __expf(a1 - mn);
      float rs = e0 + e1;
      for (int d = 8; d >= 1; d >>= 1) rs += __shfl_xor(rs, d);
      float alpha = __expf(m[r] - mn);
      l[r] = l[r] * alpha + rs;
      m[r] = mn;
      for (int t = 0; t < 4; ++t) acc[t][r] *= alpha;
      p0[r] = e0;
      p1[r] = e1;
    }

    // C-layout P -> wave-private LDS -> A-layout fragment (in-order LDS).
    for (int r = 0; r < 8; ++r) {
      Ps[w][r + hi * 8][l15] = (_Float16)p0[r];
      Ps[w][r + hi * 8][16 + l15] = (_Float16)p1[r];
    }
    v16h pf = ldsA(&Ps[w][l15][hi ? 16 : 0], &Ps[w][l15][(hi ? 16 : 0) + 8]);
    for (int t = 0; t < 4; ++t) {
      v16h vf = ldsB16(&Vt[t * 16 + l15][hi ? 16 : 0]);
      acc[t] = __builtin_amdgcn_wmma_f32_16x16x32_f16(false, pf, false, vf,
                                                      (short)0, acc[t], false, false);
    }
    __syncthreads();
  }

  // Normalize and write [B, N, D] with col = h*64 + hd.
  for (int t = 0; t < 4; ++t) {
    for (int r = 0; r < 8; ++r) {
      int n = qRow + r + hi * 8;
      int col = h * HD_ + t * 16 + l15;
      attn[((size_t)b * N_ + n) * D_ + col] = acc[t][r] / l[r];
    }
  }
}

// ---------------------------------------------------------------------------
extern "C" void kernel_launch(void* const* d_in, const int* in_sizes, int n_in,
                              void* d_out, int out_size, void* d_ws, size_t ws_size,
                              hipStream_t stream) {
  const float* x = (const float*)d_in[0];
  const float* Wq = (const float*)d_in[1];
  const float* bq = (const float*)d_in[2];
  const float* Wk = (const float*)d_in[3];
  const float* bk = (const float*)d_in[4];
  const float* Wv = (const float*)d_in[5];
  const float* bv = (const float*)d_in[6];
  const float* Wo = (const float*)d_in[7];
  const float* bo = (const float*)d_in[8];
  float* out = (float*)d_out;

  float* ws = (float*)d_ws;
  float* qb = ws;             // f32 [B,H,N,HD]
  float* kb = qb + SZ_;       // f32 [B,H,N,HD]
  float* attn = kb + SZ_;     // f32 [B,N,D]
  float* stats = attn + SZ_;  // f32 [2,B,H,HD,2]
  _Float16* kh = (_Float16*)(stats + 16384);  // f16 [B,H,N,HD]
  _Float16* vth = kh + SZ_;                   // f16 [B,H,HD,N]

  dim3 gemmGrid(D_ / 128, (B_ * N_) / 128);
  gemm_f16_wmma<<<gemmGrid, 256, 0, stream>>>(x, Wq, bq, nullptr, qb, nullptr, 1);
  gemm_f16_wmma<<<gemmGrid, 256, 0, stream>>>(x, Wk, bk, nullptr, kb, nullptr, 1);
  gemm_f16_wmma<<<gemmGrid, 256, 0, stream>>>(x, Wv, bv, nullptr, nullptr, vth, 2);

  adain_stats<<<dim3(B_ * H_, 2), 256, 0, stream>>>(qb, kb, stats);
  adain_apply<<<8192, 256, 0, stream>>>(qb, kb, kh, stats);

  flash_attn<<<dim3(N_ / 128, B_ * H_), 256, 0, stream>>>(qb, kh, vth, attn);

  gemm_f16_wmma<<<gemmGrid, 256, 0, stream>>>(attn, Wo, bo, x, out, nullptr, 0);
}
